// ModelNet40_PT3d_57028575756375
// MI455X (gfx1250) — matrix operand order repaired and checked by reading, not compiled, not run on hardware
//
#include <hip/hip_runtime.h>
#include <hip/hip_bf16.h>
#include <stdint.h>

typedef __attribute__((ext_vector_type(16))) _Float16 v16h;
typedef __attribute__((ext_vector_type(8)))  float    v8f;

#define BSZ    8
#define PTS_N  8192

// ---------------------------------------------------------------------------
// KNN: one query per thread, candidate points tiled through LDS, per-thread
// sorted insertion list of size K (ties keep earlier index, matching top_k).
// Current worst distance is cached in a register so the hot compare never
// touches the scratch-resident list; list traffic happens only on inserts.
// ---------------------------------------------------------------------------
template<int KK>
__global__ __launch_bounds__(128)
void knn_kernel(const float* __restrict__ pts, int* __restrict__ idx,
                int N, int M) {
  const int b = blockIdx.y;
  const int m = blockIdx.x * blockDim.x + threadIdx.x;
  const float* pb = pts + (size_t)b * PTS_N * 3;
  const bool active = (m < M);
  float qx = 0.f, qy = 0.f, qz = 0.f;
  if (active) { qx = pb[3*m+0]; qy = pb[3*m+1]; qz = pb[3*m+2]; }
  float dist[KK];
  int   ind[KK];
#pragma unroll
  for (int i = 0; i < KK; ++i) { dist[i] = 3.402823e38f; ind[i] = 0; }
  float worst = 3.402823e38f;
  __shared__ float tile[128 * 3];
  for (int base = 0; base < N; base += 128) {
    int cnt = N - base; if (cnt > 128) cnt = 128;
    for (int t = threadIdx.x; t < cnt * 3; t += blockDim.x)
      tile[t] = pb[base * 3 + t];
    // prefetch next candidate tile while this one is consumed
    if (base + 128 < N)
      __builtin_prefetch(&pb[(base + 128) * 3 + threadIdx.x * 3], 0, 1);
    __syncthreads();
    if (active) {
      for (int j = 0; j < cnt; ++j) {
        float dx = tile[3*j+0] - qx;
        float dy = tile[3*j+1] - qy;
        float dz = tile[3*j+2] - qz;
        float d2 = dx*dx + dy*dy + dz*dz;
        if (d2 < worst) {
          int pos = KK - 1;
          while (pos > 0 && dist[pos-1] > d2) {
            dist[pos] = dist[pos-1]; ind[pos] = ind[pos-1]; --pos;
          }
          dist[pos] = d2; ind[pos] = base + j;
          worst = dist[KK-1];
        }
      }
    }
    __syncthreads();
  }
  if (active) {
    int* op = idx + ((size_t)b * M + m) * KK;
#pragma unroll
    for (int i = 0; i < KK; ++i) op[i] = ind[i];
  }
}

// ---------------------------------------------------------------------------
// Fused weight-MLP + neighborhood aggregation.
// One block per output row (b,m).  Phase A: stage MLP params in LDS.
// Phase B: lanes t<K run the 48->32->16->16 MLP for neighbor t -> d in LDS.
// Phase C: thread c computes agg[row, c*16+n] = sum_k x[b,idx[k],c]*d[k][n],
// stored directly as f16 (WMMA A operand).  Pad channels write zeros.
// ---------------------------------------------------------------------------
#define SW_C0   0
#define SW_L1W  48
#define SW_L1B  1584
#define SW_L2W  1616
#define SW_L2B  2128
#define SW_L3W  2144
#define SW_L3B  2400
#define SW_TOT  2416

template<int KK>
__global__ __launch_bounds__(128)
void agg_kernel(const float* __restrict__ x,      // [B, N, Cin] f32
                const float* __restrict__ pts,    // [B, PTS_N, 3]
                const int*   __restrict__ idx,    // [B, M, KK]
                const float* __restrict__ centers,
                const float* __restrict__ l1w, const float* __restrict__ l1b,
                const float* __restrict__ l2w, const float* __restrict__ l2b,
                const float* __restrict__ l3w, const float* __restrict__ l3b,
                _Float16*    __restrict__ agg,    // [B*M, KP] f16
                int N, int M, int Cin, int KP) {
  const int row = blockIdx.x;
  const int b = row / M;
  const int m = row - b * M;
  const int t = threadIdx.x;

  __shared__ float sw[SW_TOT];
  __shared__ float dl[KK * 16];
  __shared__ int   il[KK];

  for (int i = t; i < 48;   i += 128) sw[SW_C0  + i] = centers[i];
  for (int i = t; i < 1536; i += 128) sw[SW_L1W + i] = l1w[i];
  for (int i = t; i < 32;   i += 128) sw[SW_L1B + i] = l1b[i];
  for (int i = t; i < 512;  i += 128) sw[SW_L2W + i] = l2w[i];
  for (int i = t; i < 16;   i += 128) sw[SW_L2B + i] = l2b[i];
  for (int i = t; i < 256;  i += 128) sw[SW_L3W + i] = l3w[i];
  for (int i = t; i < 16;   i += 128) sw[SW_L3B + i] = l3b[i];
  if (t < KK) il[t] = idx[((size_t)row) * KK + t];
  __syncthreads();

  if (t < KK) {
    const float* pb = pts + (size_t)b * PTS_N * 3;
    const int j = il[t];
    float r[3];
    r[0] = pb[3*j+0] - pb[3*m+0];
    r[1] = pb[3*j+1] - pb[3*m+1];
    r[2] = pb[3*j+2] - pb[3*m+2];
    float dv[48];
#pragma unroll
    for (int dd = 0; dd < 3; ++dd)
#pragma unroll
      for (int n = 0; n < 16; ++n)
        dv[dd*16+n] = r[dd] - sw[SW_C0 + dd*16 + n];
    float h1[32];
#pragma unroll
    for (int o = 0; o < 32; ++o) {
      float a = sw[SW_L1B + o];
#pragma unroll
      for (int d = 0; d < 48; ++d) a += dv[d] * sw[SW_L1W + d*32 + o];
      h1[o] = a > 0.f ? a : 0.f;
    }
    float h2[16];
#pragma unroll
    for (int o = 0; o < 16; ++o) {
      float a = sw[SW_L2B + o];
#pragma unroll
      for (int d = 0; d < 32; ++d) a += h1[d] * sw[SW_L2W + d*16 + o];
      h2[o] = a > 0.f ? a : 0.f;
    }
#pragma unroll
    for (int o = 0; o < 16; ++o) {
      float a = sw[SW_L3B + o];
#pragma unroll
      for (int d = 0; d < 16; ++d) a += h2[d] * sw[SW_L3W + d*16 + o];
      dl[t*16 + o] = a > 0.f ? a : 0.f;
    }
  }
  __syncthreads();

  const int CP = KP >> 4;
  for (int c = t; c < CP; c += 128) {
    float acc[16];
#pragma unroll
    for (int n = 0; n < 16; ++n) acc[n] = 0.f;
    if (c < Cin) {
      for (int k = 0; k < KK; ++k) {
        float xv = x[((size_t)b * N + il[k]) * Cin + c];
#pragma unroll
        for (int n = 0; n < 16; ++n) acc[n] += xv * dl[k*16 + n];
      }
    }
    _Float16* dst = agg + (size_t)row * KP + c * 16;
#pragma unroll
    for (int n = 0; n < 16; ++n) dst[n] = (_Float16)acc[n];
  }
}

// ---------------------------------------------------------------------------
// Convert f32 weight [Kdim, Nc] -> transposed, K-padded f16 [Nc, KP].
// ---------------------------------------------------------------------------
__global__ void wconvert_kernel(const float* __restrict__ w, _Float16* __restrict__ wT,
                                int Kdim, int KP, int Nc) {
  int i = blockIdx.x * blockDim.x + threadIdx.x;
  if (i >= Nc * KP) return;
  int n = i / KP;
  int k = i - n * KP;
  wT[i] = (k < Kdim) ? (_Float16)w[(size_t)k * Nc + n] : (_Float16)0.f;
}

// ---------------------------------------------------------------------------
// WMMA GEMM: Y[R,Nc] = (A[R,KP] @ B[KP,Nc]) * scale + bias, f16 in / f32 out.
// One wave computes a 16x64 output tile (4 accumulators): each A fragment is
// loaded once and feeds 4 v_wmma issues, 4x less A-side traffic.
// Fragments follow ISA 7.12.2 layouts:
//   A 16x32 f16: lane<16 -> K {0..7,16..23}, lane>=16 -> K {8..15,24..31}
//   B 32x16 f16 (BT row-major [Nc,KP]): lane<16 -> K {0..7,16..23}, lane>=16 -> K {8..15,24..31}
//   C/D: vgpr j -> M = j (+8 for upper half-wave), N = lane&15
// Requires Nc % 64 == 0 (true for all layers: 64,128,256,256,512).
// ---------------------------------------------------------------------------
__global__ __launch_bounds__(128)
void gemm_wmma_kernel(const _Float16* __restrict__ A, const _Float16* __restrict__ BT,
                      const float* __restrict__ bias, float* __restrict__ Y,
                      int R, int KP, int Nc, float scale) {
  const int lane = threadIdx.x & 31;
  const int wave = threadIdx.x >> 5;
  const int tiles_n4 = Nc >> 6;                  // 16x64 tiles along N
  const int total = (R >> 4) * tiles_n4;
  const int tile = blockIdx.x * (blockDim.x >> 5) + wave;
  if (tile >= total) return;
  const int tm  = tile / tiles_n4;
  const int tn4 = tile - tm * tiles_n4;          // 64-wide N tile
  const int l15 = lane & 15;
  const int hi  = (lane & 16) ? 1 : 0;

  const uint32_t* Ap = (const uint32_t*)(A + (size_t)(tm*16 + l15) * KP);
  const uint32_t* Bp[4];
#pragma unroll
  for (int q = 0; q < 4; ++q)
    Bp[q] = (const uint32_t*)(BT + (size_t)(tn4*64 + q*16 + l15) * KP);

  v8f c0 = {0.f,0.f,0.f,0.f,0.f,0.f,0.f,0.f};
  v8f c1 = c0, c2 = c0, c3 = c0;

  for (int kt = 0; kt < KP; kt += 32) {
    union { uint32_t u[8]; v16h v; } af, bf0, bf1, bf2, bf3;
    const int ka = (kt + (hi ? 8 : 0)) >> 1;      // dword index into A row
#pragma unroll
    for (int j = 0; j < 4; ++j) {
      af.u[j]     = Ap[ka + j];                   // K = kb+2j, kb+2j+1
      af.u[4 + j] = Ap[ka + 8 + j];               // K = kb+16+2j, ...
    }
    const int kb = (kt + (hi ? 8 : 0)) >> 1;      // dword index into B col
#pragma unroll
    for (int j = 0; j < 8; ++j) {
      bf0.u[j] = Bp[0][kb + j + (j & 4)];
      bf1.u[j] = Bp[1][kb + j + (j & 4)];
      bf2.u[j] = Bp[2][kb + j + (j & 4)];
      bf3.u[j] = Bp[3][kb + j + (j & 4)];
    }
    c0 = __builtin_amdgcn_wmma_f32_16x16x32_f16(false, af.v, false, bf0.v,
                                                (short)0, c0, false, false);
    c1 = __builtin_amdgcn_wmma_f32_16x16x32_f16(false, af.v, false, bf1.v,
                                                (short)0, c1, false, false);
    c2 = __builtin_amdgcn_wmma_f32_16x16x32_f16(false, af.v, false, bf2.v,
                                                (short)0, c2, false, false);
    c3 = __builtin_amdgcn_wmma_f32_16x16x32_f16(false, af.v, false, bf3.v,
                                                (short)0, c3, false, false);
  }

  const int mbase = tm * 16 + (hi ? 8 : 0);
#pragma unroll
  for (int q = 0; q < 4; ++q) {
    const int col = tn4*64 + q*16 + l15;
    const float bv = bias[col];
    const v8f cq = (q == 0) ? c0 : (q == 1) ? c1 : (q == 2) ? c2 : c3;
#pragma unroll
    for (int j = 0; j < 8; ++j)
      Y[(size_t)(mbase + j) * Nc + col] = cq[j] * scale + bv;
  }
}

// ---------------------------------------------------------------------------
// GroupNorm: per-(batch,group) mean / rsqrt(var+eps) over (points, chans).
// ---------------------------------------------------------------------------
__global__ __launch_bounds__(256)
void gn_stats_kernel(const float* __restrict__ Y, float* __restrict__ stats,
                     int M, int C, int cg) {
  const int g = blockIdx.x, b = blockIdx.y, G = gridDim.x;
  const int total = M * cg;
  float s = 0.f, s2 = 0.f;
  for (int i = threadIdx.x; i < total; i += blockDim.x) {
    int m = i / cg, cc = i - m * cg;
    float v = Y[((size_t)b * M + m) * C + g * cg + cc];
    s += v; s2 += v * v;
  }
  __shared__ float sh[256], sh2[256];
  sh[threadIdx.x] = s; sh2[threadIdx.x] = s2;
  __syncthreads();
  for (int step = 128; step > 0; step >>= 1) {
    if (threadIdx.x < step) {
      sh[threadIdx.x]  += sh[threadIdx.x + step];
      sh2[threadIdx.x] += sh2[threadIdx.x + step];
    }
    __syncthreads();
  }
  if (threadIdx.x == 0) {
    float inv  = 1.f / (float)total;
    float mean = sh[0] * inv;
    float var  = sh2[0] * inv - mean * mean;
    stats[((size_t)b * G + g) * 2 + 0] = mean;
    stats[((size_t)b * G + g) * 2 + 1] = rsqrtf(var + 1e-5f);
  }
}

__global__ void gn_apply_kernel(const float* __restrict__ Y, const float* __restrict__ stats,
                                const float* __restrict__ gamma, const float* __restrict__ beta,
                                float* __restrict__ Xo, int M, int C, int cg, int G, int total) {
  int i = blockIdx.x * blockDim.x + threadIdx.x;
  if (i >= total) return;
  int c  = i % C;
  int b  = i / (M * C);
  int g  = c / cg;
  float mean = stats[((size_t)b * G + g) * 2 + 0];
  float rstd = stats[((size_t)b * G + g) * 2 + 1];
  float v = (Y[i] - mean) * rstd * gamma[c] + beta[c];
  Xo[i] = v > 0.f ? v : 0.f;
}

// ---------------------------------------------------------------------------
// Head: mean-pool over 128 points -> 512 -> 128 -> 40.  One block per batch.
// ---------------------------------------------------------------------------
__global__ __launch_bounds__(128)
void head_kernel(const float* __restrict__ x5,
                 const float* __restrict__ w1, const float* __restrict__ b1,
                 const float* __restrict__ w2, const float* __restrict__ b2,
                 float* __restrict__ out) {
  const int b = blockIdx.x;
  const int t = threadIdx.x;
  __shared__ float pooled[512];
  __shared__ float h[128];
  for (int c = t; c < 512; c += 128) {
    float s = 0.f;
    for (int m = 0; m < 128; ++m)
      s += x5[((size_t)b * 128 + m) * 512 + c];
    pooled[c] = s * (1.f / 128.f);
  }
  __syncthreads();
  {
    float a = b1[t];
    for (int c = 0; c < 512; ++c) a += pooled[c] * w1[c * 128 + t];
    h[t] = a;
  }
  __syncthreads();
  if (t < 40) {
    float a = b2[t];
    for (int c = 0; c < 128; ++c) a += h[c] * w2[c * 40 + t];
    out[b * 40 + t] = a;
  }
}

// ---------------------------------------------------------------------------
// Host driver
// ---------------------------------------------------------------------------
extern "C" void kernel_launch(void* const* d_in, const int* in_sizes, int n_in,
                              void* d_out, int out_size, void* d_ws, size_t ws_size,
                              hipStream_t stream) {
  (void)in_sizes; (void)n_in; (void)out_size; (void)ws_size;

  const float* xin  = (const float*)d_in[0];   // [8, 8192, 3]
  const float* pts  = (const float*)d_in[1];   // [8, 8192, 3]
  auto in = [&](int i) { return (const float*)d_in[i]; };

  // workspace carve-up (256B aligned)
  char* ws = (char*)d_ws;
  size_t off = 0;
  auto carve = [&](size_t bytes) {
    void* p = ws + off;
    off = (off + bytes + 255) & ~(size_t)255;
    return p;
  };
  float*    ybuf  = (float*)   carve((size_t)2097152 * 4);   // max R*Co
  float*    x1    = (float*)   carve((size_t)2097152 * 4);   // 8*4096*64
  float*    x2    = (float*)   carve((size_t)1048576 * 4);   // 8*1024*128
  float*    x3    = (float*)   carve((size_t)1048576 * 4);   // 8*512*256
  float*    x4    = (float*)   carve((size_t)524288  * 4);   // 8*256*256
  float*    x5    = (float*)   carve((size_t)524288  * 4);   // 8*128*512
  _Float16* agg   = (_Float16*)carve((size_t)8388608 * 2);   // max R*KP
  _Float16* wT    = (_Float16*)carve((size_t)2097152 * 2);   // max Co*KP
  int*      idxb  = (int*)     carve((size_t)1048576 * 4);   // max B*M*K
  float*    stats = (float*)   carve((size_t)4096    * 4);   // B*Gmax*2

  const int Ns[6] = {8192, 4096, 1024, 512, 256, 128};
  const int Ks[5] = {32, 32, 16, 16, 16};
  const int Ci[5] = {3, 64, 128, 256, 256};
  const int Co[5] = {64, 128, 256, 256, 512};
  const int Gs[5] = {32, 64, 128, 128, 256};
  const int KPs[5] = {64, 1024, 2048, 4096, 4096};
  float* xouts[5] = {x1, x2, x3, x4, x5};

  for (int L = 0; L < 5; ++L) {
    const int N  = Ns[L], M = Ns[L+1], K = Ks[L];
    const int ci = Ci[L], co = Co[L], G = Gs[L], KP = KPs[L];
    const int R  = BSZ * M;
    const int base = 2 + L * 11;
    const float* wgt = in(base + 0);
    const float* bia = in(base + 1);
    const float* cen = in(base + 2);
    const float* l1w = in(base + 3); const float* l1b = in(base + 4);
    const float* l2w = in(base + 5); const float* l2b = in(base + 6);
    const float* l3w = in(base + 7); const float* l3b = in(base + 8);
    const float* gng = in(base + 9); const float* gnb = in(base + 10);

    dim3 kg((M + 127) / 128, BSZ);
    if (K == 32) knn_kernel<32><<<kg, 128, 0, stream>>>(pts, idxb, N, M);
    else         knn_kernel<16><<<kg, 128, 0, stream>>>(pts, idxb, N, M);

    if (K == 32)
      agg_kernel<32><<<R, 128, 0, stream>>>(xin, pts, idxb, cen,
                                            l1w, l1b, l2w, l2b, l3w, l3b,
                                            agg, N, M, ci, KP);
    else
      agg_kernel<16><<<R, 128, 0, stream>>>(xin, pts, idxb, cen,
                                            l1w, l1b, l2w, l2b, l3w, l3b,
                                            agg, N, M, ci, KP);

    const int kd = ci * 16;
    wconvert_kernel<<<(co * KP + 255) / 256, 256, 0, stream>>>(wgt, wT, kd, KP, co);

    const int tiles = (R / 16) * (co / 64);
    gemm_wmma_kernel<<<(tiles + 3) / 4, 128, 0, stream>>>(agg, wT, bia, ybuf,
                                                          R, KP, co, 1.0f / (float)K);

    gn_stats_kernel<<<dim3(G, BSZ), 256, 0, stream>>>(ybuf, stats, M, co, co / G);
    const int tot = R * co;
    gn_apply_kernel<<<(tot + 255) / 256, 256, 0, stream>>>(ybuf, stats, gng, gnb,
                                                           xouts[L], M, co, co / G, G, tot);
    xin = xouts[L];
  }

  head_kernel<<<BSZ, 128, 0, stream>>>(x5, in(57), in(58), in(59), in(60), (float*)d_out);
}
